// Attn_40029095198891
// MI455X (gfx1250) — matrix operand order repaired and checked by reading, not compiled, not run on hardware
//
#include <hip/hip_runtime.h>
#include <hip/hip_bf16.h>
#include <math.h>

typedef __attribute__((ext_vector_type(16))) _Float16 v16h;
typedef __attribute__((ext_vector_type(8)))  _Float16 v8h;
typedef __attribute__((ext_vector_type(8)))  float    v8f;

#define CB 2
#define CT 1024
#define CC 1024
#define CH 16
#define CNOPE 32
#define CVH 32
#define CRD 64
#define CQD 96          // NOPE + ROPE_D
#define CKEEP 256
#define CQLORA 96
#define CKVLORA 32
#define LN1E4 9.210340371976184f

// ---------------------------------------------------------------------------
// WMMA fragment helpers (wave32, 16x16x32 f16 -> f32)
// A layout (16x32, MxK): m = lane&15, half = lane>>4,
//   element e : K = half*16 + e
//   -> two contiguous 8-half (16B) runs: vectorized ds/global loads.
// B layout (32x16, KxN): n = lane&15, element e : K = half*16 + e
//   -> for transposed sources (X[n][k]) one contiguous 16-half run.
// C/D layout (16x16): n = lane&15, row m = v + 8*half
// ---------------------------------------------------------------------------
__device__ __forceinline__ v16h load_frag_a(const _Float16* p, int ld) {
  int lane = threadIdx.x & 31, m = lane & 15, half = lane >> 4;
  const _Float16* base = p + m * ld + half * 16;
  v8h lo = *(const v8h*)(base);
  v8h hi = *(const v8h*)(base + 8);
  v16h a;
#pragma unroll
  for (int e = 0; e < 8; ++e) { a[e] = lo[e]; a[8 + e] = hi[e]; }
  return a;
}

// B[k][n] = X[n][k]  (transposed source: X row-major NxK, e.g. K-matrix rows)
__device__ __forceinline__ v16h load_frag_bT(const _Float16* p, int ld) {
  int lane = threadIdx.x & 31, n = lane & 15, half = lane >> 4;
  const _Float16* base = p + n * ld + half * 16;
  v8h lo = *(const v8h*)(base);
  v8h hi = *(const v8h*)(base + 8);
  v16h b;
#pragma unroll
  for (int e = 0; e < 8; ++e) { b[e] = lo[e]; b[8 + e] = hi[e]; }
  return b;
}

// B[k][n] from row-major KxN storage (strided gather; used only for V tiles)
__device__ __forceinline__ v16h load_frag_b(const _Float16* p, int ld) {
  int lane = threadIdx.x & 31, n = lane & 15, half = lane >> 4;
  v16h b;
#pragma unroll
  for (int e = 0; e < 16; ++e) b[e] = p[(half * 16 + e) * ld + n];
  return b;
}

__device__ __forceinline__ v8f wmma16(v16h a, v16h b, v8f c) {
  return __builtin_amdgcn_wmma_f32_16x16x32_f16(false, a, false, b, (short)0, c,
                                                false, false);
}

__device__ __forceinline__ void rope_pair(int pos, int j, float x1, float x2,
                                          float& o1, float& o2) {
  float f = __expf(-((float)(2 * j) / 64.0f) * LN1E4);
  float sn, cs;
  __sincosf((float)pos * f, &sn, &cs);
  o1 = x1 * cs - x2 * sn;
  o2 = x1 * sn + x2 * cs;
}

// ---------------------------------------------------------------------------
// 1. per-token small projections + RMS norms + f16 copy of x
// blockDim = 160, grid = B*T
// ---------------------------------------------------------------------------
__global__ void smallproj_kernel(const float* __restrict__ x,
                                 const float* __restrict__ w_cq,
                                 const float* __restrict__ g_q,
                                 const float* __restrict__ w_ckv,
                                 const float* __restrict__ g_kv,
                                 const float* __restrict__ w_imp,
                                 const float* __restrict__ w_gate,
                                 _Float16* __restrict__ x_h,
                                 _Float16* __restrict__ nq_h,
                                 _Float16* __restrict__ ckv_h,
                                 float* __restrict__ scores,
                                 float* __restrict__ glog) {
  __shared__ float xs[CC];
  __shared__ float tq[CQLORA];
  __shared__ float tkv[CKVLORA];
  __shared__ float rs[2];
  int bt = blockIdx.x;
  int tid = threadIdx.x;
  const float* xr = x + (size_t)bt * CC;
  for (int c = tid; c < CC; c += 160) {
    float v = xr[c];
    xs[c] = v;
    x_h[(size_t)bt * CC + c] = (_Float16)v;
  }
  __syncthreads();
  if (tid < CQLORA) {
    float acc = 0.f;
    for (int c = 0; c < CC; ++c) acc += xs[c] * w_cq[c * CQLORA + tid];
    tq[tid] = acc;
  } else if (tid < CQLORA + CKVLORA) {
    int j = tid - CQLORA;
    float acc = 0.f;
    for (int c = 0; c < CC; ++c) acc += xs[c] * w_ckv[c * CKVLORA + j];
    tkv[j] = acc;
  } else if (tid == 128) {
    float acc = 0.f;
    for (int c = 0; c < CC; ++c) acc += xs[c] * w_imp[c];
    scores[bt] = acc;
  } else if (tid >= 129 && tid < 132) {
    int j = tid - 129;
    float acc = 0.f;
    for (int c = 0; c < CC; ++c) acc += xs[c] * w_gate[c * 3 + j];
    glog[(size_t)bt * 3 + j] = acc;
  }
  __syncthreads();
  if (tid == 0) {
    float s = 0.f;
    for (int j = 0; j < CQLORA; ++j) s += tq[j] * tq[j];
    rs[0] = rsqrtf(s / (float)CQLORA + 1e-6f);
  }
  if (tid == 1) {
    float s = 0.f;
    for (int j = 0; j < CKVLORA; ++j) s += tkv[j] * tkv[j];
    rs[1] = rsqrtf(s / (float)CKVLORA + 1e-6f);
  }
  __syncthreads();
  if (tid < CQLORA) {
    nq_h[(size_t)bt * CQLORA + tid] = (_Float16)(tq[tid] * rs[0] * g_q[tid]);
  } else if (tid < CQLORA + CKVLORA) {
    int j = tid - CQLORA;
    ckv_h[(size_t)bt * CKVLORA + j] = (_Float16)(tkv[j] * rs[1] * g_kv[j]);
  }
}

// ---------------------------------------------------------------------------
// 2. gate = softmax(mean_t(glog)) per batch.  grid = B, blockDim = 256
// ---------------------------------------------------------------------------
__global__ void gate_kernel(const float* __restrict__ glog,
                            float* __restrict__ gate, int T) {
  __shared__ float sm[3 * 256];
  int b = blockIdx.x, tid = threadIdx.x;
  float a0 = 0.f, a1 = 0.f, a2 = 0.f;
  for (int t = tid; t < T; t += 256) {
    const float* p = glog + ((size_t)b * T + t) * 3;
    a0 += p[0]; a1 += p[1]; a2 += p[2];
  }
  sm[tid] = a0; sm[256 + tid] = a1; sm[512 + tid] = a2;
  __syncthreads();
  for (int s = 128; s > 0; s >>= 1) {
    if (tid < s) {
      sm[tid] += sm[tid + s];
      sm[256 + tid] += sm[256 + tid + s];
      sm[512 + tid] += sm[512 + tid + s];
    }
    __syncthreads();
  }
  if (tid == 0) {
    float m0 = sm[0] / (float)T, m1 = sm[256] / (float)T, m2 = sm[512] / (float)T;
    float mx = fmaxf(m0, fmaxf(m1, m2));
    float e0 = __expf(m0 - mx), e1 = __expf(m1 - mx), e2 = __expf(m2 - mx);
    float s = e0 + e1 + e2;
    gate[b * 3 + 0] = e0 / s;
    gate[b * 3 + 1] = e1 / s;
    gate[b * 3 + 2] = e2 / s;
  }
}

// ---------------------------------------------------------------------------
// 3. exact top-K by rank (ties -> lower index), output index-sorted ascending
// grid = B, blockDim = T (1024)
// ---------------------------------------------------------------------------
__global__ void topk_kernel(const float* __restrict__ scores,
                            int* __restrict__ idx, int T) {
  __shared__ float s[CT];
  __shared__ int flag[CT];
  int b = blockIdx.x, t = threadIdx.x;
  s[t] = scores[(size_t)b * T + t];
  __syncthreads();
  float my = s[t];
  int rank = 0;
  for (int j = 0; j < T; ++j) {
    float o = s[j];
    rank += (o > my) || (o == my && j < t);
  }
  flag[t] = (rank < CKEEP) ? 1 : 0;
  __syncthreads();
  if (flag[t]) {
    int pos = 0;
    for (int j = 0; j < t; ++j) pos += flag[j];
    idx[b * CKEEP + pos] = t;
  }
}

__global__ void gather_kernel(const float* __restrict__ x,
                              const int* __restrict__ idx,
                              _Float16* __restrict__ sel_h) {
  int bs = blockIdx.x;
  int b = bs / CKEEP;
  int t = idx[bs];
  const float* src = x + ((size_t)b * CT + t) * CC;
  _Float16* dst = sel_h + (size_t)bs * CC;
  for (int c = threadIdx.x; c < CC; c += blockDim.x) dst[c] = (_Float16)src[c];
}

// ---------------------------------------------------------------------------
// 4. WMMA GEMM: C[M,N] = scale * (A_f16[M,K] @ B_f32[K,N])
// block = 128 threads (4 waves), 64x64 output tile:
//   wave w computes rows [tmb+16w, tmb+16w+16), all 64 columns (4 wmma/step).
// A and B staged in LDS *pre-swizzled into fragment lines* so every fragment
// read is two ds_load_b128. A staging itself is two 16B global loads/thread.
// grid = (N/64, M/64)
// ---------------------------------------------------------------------------
__global__ void gemm_wmma_kernel(const _Float16* __restrict__ A, int lda,
                                 const float* __restrict__ Bw, int ldb,
                                 float* __restrict__ Cm, int ldc, int K,
                                 float scale) {
  __shared__ __align__(32) _Float16 As[128 * 16];  // frag line per (waveRow, lane)
  __shared__ __align__(32) _Float16 Bs[128 * 16];  // frag line per (nTile, lane)
  int tid = threadIdx.x;
  int wave = tid >> 5, lane = tid & 31;
  int n15 = lane & 15, half = lane >> 4;
  int tmb = blockIdx.y * 64, tn0 = blockIdx.x * 64;
  v8f acc0 = {}, acc1 = {}, acc2 = {}, acc3 = {};

  const _Float16* aSrc = A + (size_t)(tmb + wave * 16 + n15) * lda + half * 16;
  const float* bSrc = Bw + (size_t)(half * 16) * ldb + tn0 + wave * 16 + n15;

  for (int k0 = 0; k0 < K; k0 += 32) {
    // ---- stage A fragment lines: two contiguous 16B runs per thread ----
    {
      const _Float16* s = aSrc + k0;
      v8h lo = *(const v8h*)(s);
      v8h hi = *(const v8h*)(s + 8);
      *(v8h*)(As + tid * 16) = lo;
      *(v8h*)(As + tid * 16 + 8) = hi;
      if (k0 + 32 < K) __builtin_prefetch((const void*)(s + 32), 0, 3);
    }
    // ---- stage B fragment lines: column gather from f32 weights ----
    {
      const float* s = bSrc + (size_t)k0 * ldb;
      _Float16 tmp[16];
#pragma unroll
      for (int e = 0; e < 16; ++e) tmp[e] = (_Float16)s[(size_t)e * ldb];
#pragma unroll
      for (int e = 0; e < 16; ++e) Bs[tid * 16 + e] = tmp[e];
      if (k0 + 32 < K) __builtin_prefetch((const void*)(s + (size_t)32 * ldb), 0, 3);
    }
    __syncthreads();
    v16h a  = *(const v16h*)(As + (wave * 32 + lane) * 16);
    v16h b0 = *(const v16h*)(Bs + (0 * 32 + lane) * 16);
    v16h b1 = *(const v16h*)(Bs + (1 * 32 + lane) * 16);
    v16h b2 = *(const v16h*)(Bs + (2 * 32 + lane) * 16);
    v16h b3 = *(const v16h*)(Bs + (3 * 32 + lane) * 16);
    acc0 = wmma16(a, b0, acc0);
    acc1 = wmma16(a, b1, acc1);
    acc2 = wmma16(a, b2, acc2);
    acc3 = wmma16(a, b3, acc3);
    __syncthreads();
  }
  int tm = tmb + wave * 16;
#pragma unroll
  for (int v = 0; v < 8; ++v) {
    int mm = v + 8 * half;
    float* dst = Cm + (size_t)(tm + mm) * ldc + tn0 + n15;
    dst[0]  = acc0[v] * scale;
    dst[16] = acc1[v] * scale;
    dst[32] = acc2[v] * scale;
    dst[48] = acc3[v] * scale;
  }
}

// ---------------------------------------------------------------------------
// 5. pack kernels: rope + transpose into [B,H,S,D] f16
// ---------------------------------------------------------------------------
__global__ void pack_q_kernel(const float* __restrict__ qn,
                              const float* __restrict__ qr,
                              _Float16* __restrict__ Q) {
  int id = blockIdx.x;                  // ((b*T + t)*H + h)
  int h = id % CH;
  int bt = id / CH;
  int t = bt % CT;
  int b = bt / CT;
  _Float16* dst = Q + (((size_t)(b * CH + h) * CT) + t) * CQD;
  int tid = threadIdx.x;
  if (tid < 32) {
    dst[tid] = (_Float16)qn[(size_t)bt * (CH * CNOPE) + h * CNOPE + tid];
  } else {
    int j = tid - 32;
    const float* r = qr + (size_t)bt * (CH * CRD) + h * CRD;
    float o1, o2;
    rope_pair(t, j, r[j], r[32 + j], o1, o2);
    dst[32 + j] = (_Float16)o1;
    dst[64 + j] = (_Float16)o2;
  }
}

__global__ void pack_k1_kernel(const float* __restrict__ knope,
                               const float* __restrict__ krope,
                               const float* __restrict__ v1,
                               _Float16* __restrict__ K,
                               _Float16* __restrict__ V) {
  int id = blockIdx.x;
  int h = id % CH;
  int bt = id / CH;
  int t = bt % CT;
  int b = bt / CT;
  _Float16* kd = K + (((size_t)(b * CH + h) * CT) + t) * CQD;
  int tid = threadIdx.x;
  if (tid < 32) {
    kd[tid] = (_Float16)knope[(size_t)bt * (CH * CNOPE) + h * CNOPE + tid];
  } else if (tid < 64) {
    int j = tid - 32;
    const float* r = krope + (size_t)bt * CRD;   // already scaled by 1/N_HEAD
    float o1, o2;
    rope_pair(t, j, r[j], r[32 + j], o1, o2);
    kd[32 + j] = (_Float16)o1;
    kd[64 + j] = (_Float16)o2;
  } else {
    int d = tid - 64;
    V[(((size_t)(b * CH + h) * CT) + t) * CVH + d] =
        (_Float16)v1[(size_t)bt * (CH * CVH) + h * CVH + d];
  }
}

// used for both selected (S=KEEP, pos=s) and window (S=T, pos=t) branches
__global__ void pack_skw_kernel(const float* __restrict__ kraw,
                                const float* __restrict__ vraw,
                                _Float16* __restrict__ K,
                                _Float16* __restrict__ V, int S) {
  int id = blockIdx.x;
  int h = id % CH;
  int bs = id / CH;
  int s = bs % S;
  int b = bs / S;
  const float* kr = kraw + (size_t)bs * (CH * CQD) + h * CQD;
  _Float16* kd = K + (((size_t)(b * CH + h) * S) + s) * CQD;
  int tid = threadIdx.x;
  if (tid < 32) {
    kd[tid] = (_Float16)kr[tid];
  } else if (tid < 64) {
    int j = tid - 32;
    float o1, o2;
    rope_pair(s, j, kr[32 + j], kr[64 + j], o1, o2);
    kd[32 + j] = (_Float16)o1;
    kd[64 + j] = (_Float16)o2;
  } else {
    int d = tid - 64;
    V[(((size_t)(b * CH + h) * S) + s) * CVH + d] =
        (_Float16)vraw[(size_t)bs * (CH * CVH) + h * CVH + d];
  }
}

// ---------------------------------------------------------------------------
// 6. flash attention: one wave per (b, h, 16-query tile); online softmax
// grid = (Tq/16, H, B), blockDim = 32
// ---------------------------------------------------------------------------
__global__ void attn_kernel(const _Float16* __restrict__ Q,
                            const _Float16* __restrict__ K,
                            const _Float16* __restrict__ V,
                            float* __restrict__ O, int Tq, int Tk, int causal) {
  __shared__ __align__(32) _Float16 Qs[16 * CQD];
  __shared__ __align__(32) _Float16 Ks[32 * CQD];
  __shared__ __align__(32) _Float16 Vs[32 * CVH];
  __shared__ __align__(32) _Float16 Ps[16 * 32];
  int qt = blockIdx.x, h = blockIdx.y, b = blockIdx.z;
  int H = gridDim.y;
  int lane = threadIdx.x;
  int q0 = qt * 16;
  const _Float16* Qb = Q + (((size_t)(b * H + h) * Tq) + q0) * CQD;
  const _Float16* Kb = K + ((size_t)(b * H + h) * Tk) * CQD;
  const _Float16* Vb = V + ((size_t)(b * H + h) * Tk) * CVH;
  for (int i = lane; i < 16 * CQD; i += 32) Qs[i] = Qb[i];
  __syncthreads();
  v16h qf0 = load_frag_a(Qs + 0, CQD);
  v16h qf1 = load_frag_a(Qs + 32, CQD);
  v16h qf2 = load_frag_a(Qs + 64, CQD);

  v8f o0 = {}, o1 = {};
  float mrow[8], lrow[8], alpha[8];
#pragma unroll
  for (int v = 0; v < 8; ++v) { mrow[v] = -1e30f; lrow[v] = 0.f; }
  int n = lane & 15, half = lane >> 4;
  int kmax = causal ? min(Tk, q0 + 16) : Tk;
  const float scale = 0.10206207261596575f;  // 1/sqrt(96)

  for (int c0 = 0; c0 < kmax; c0 += 32) {
    int valid = min(32, kmax - c0);
    for (int i = lane; i < 32 * CQD; i += 32) {
      int r = i / CQD;
      Ks[i] = (r < valid) ? Kb[(size_t)(c0 + r) * CQD + (i % CQD)]
                          : (_Float16)0.f;
    }
    for (int i = lane; i < 32 * CVH; i += 32) {
      int r = i >> 5;
      Vs[i] = (r < valid) ? Vb[(size_t)(c0 + r) * CVH + (i & 31)]
                          : (_Float16)0.f;
    }
    if (c0 + 32 < kmax)
      __builtin_prefetch((const void*)(Kb + (size_t)(c0 + 32) * CQD + lane * 4),
                         0, 3);
    __syncthreads();

    v8f S0 = {}, S1 = {};
    S0 = wmma16(qf0, load_frag_bT(Ks + 0, CQD), S0);
    S0 = wmma16(qf1, load_frag_bT(Ks + 32, CQD), S0);
    S0 = wmma16(qf2, load_frag_bT(Ks + 64, CQD), S0);
    S1 = wmma16(qf0, load_frag_bT(Ks + 16 * CQD + 0, CQD), S1);
    S1 = wmma16(qf1, load_frag_bT(Ks + 16 * CQD + 32, CQD), S1);
    S1 = wmma16(qf2, load_frag_bT(Ks + 16 * CQD + 64, CQD), S1);

#pragma unroll
    for (int v = 0; v < 8; ++v) {
      int m = v + 8 * half;
      int qidx = q0 + m;
      int k0i = c0 + n, k1i = c0 + 16 + n;
      float s0 = S0[v] * scale, s1 = S1[v] * scale;
      if (k0i >= kmax || (causal && k0i > qidx)) s0 = -1e30f;
      if (k1i >= kmax || (causal && k1i > qidx)) s1 = -1e30f;
      float mx = fmaxf(s0, s1);
      for (int off = 1; off < 16; off <<= 1) mx = fmaxf(mx, __shfl_xor(mx, off));
      float mnew = fmaxf(mrow[v], mx);
      alpha[v] = __expf(mrow[v] - mnew);
      mrow[v] = mnew;
      float e0 = __expf(s0 - mnew), e1 = __expf(s1 - mnew);
      float rs2 = e0 + e1;
      for (int off = 1; off < 16; off <<= 1) rs2 += __shfl_xor(rs2, off);
      lrow[v] = lrow[v] * alpha[v] + rs2;
      Ps[m * 32 + n] = (_Float16)e0;
      Ps[m * 32 + 16 + n] = (_Float16)e1;
    }
    __syncthreads();

    v16h pf = load_frag_a(Ps, 32);
    v16h vf0 = load_frag_b(Vs + 0, CVH);
    v16h vf1 = load_frag_b(Vs + 16, CVH);
#pragma unroll
    for (int v = 0; v < 8; ++v) { o0[v] *= alpha[v]; o1[v] *= alpha[v]; }
    o0 = wmma16(pf, vf0, o0);
    o1 = wmma16(pf, vf1, o1);
    __syncthreads();
  }

  float* Ob = O + (((size_t)(b * H + h) * Tq) + q0) * CVH;
#pragma unroll
  for (int v = 0; v < 8; ++v) {
    int m = v + 8 * half;
    float inv = 1.0f / fmaxf(lrow[v], 1e-20f);
    Ob[m * CVH + n] = o0[v] * inv;
    Ob[m * CVH + 16 + n] = o1[v] * inv;
  }
}

// ---------------------------------------------------------------------------
// 7. gated combine + transpose [B,H,T,VH] -> [B,T,H*VH] (f16 for final GEMM)
// grid = B*T, blockDim = 512
// ---------------------------------------------------------------------------
__global__ void combine_kernel(const float* __restrict__ O1,
                               const float* __restrict__ O2,
                               const float* __restrict__ O3,
                               const float* __restrict__ gate,
                               _Float16* __restrict__ attn_h) {
  int bt = blockIdx.x;
  int b = bt / CT;
  int t = bt % CT;
  int j = threadIdx.x;
  int h = j >> 5, d = j & 31;
  size_t src = (((size_t)(b * CH + h) * CT) + t) * CVH + d;
  float g0 = gate[b * 3 + 0], g1 = gate[b * 3 + 1], g2 = gate[b * 3 + 2];
  float val = g0 * O1[src] + g1 * O2[src] + g2 * O3[src];
  attn_h[(size_t)bt * (CH * CVH) + j] = (_Float16)val;
}

// ---------------------------------------------------------------------------
// host launcher
// ---------------------------------------------------------------------------
extern "C" void kernel_launch(void* const* d_in, const int* in_sizes, int n_in,
                              void* d_out, int out_size, void* d_ws,
                              size_t ws_size, hipStream_t stream) {
  const float* x        = (const float*)d_in[0];
  const float* w_cq     = (const float*)d_in[1];
  const float* g_qnorm  = (const float*)d_in[2];
  const float* w_dqn    = (const float*)d_in[3];
  const float* w_dqr    = (const float*)d_in[4];
  const float* w_ckv    = (const float*)d_in[5];
  const float* g_kvnorm = (const float*)d_in[6];
  const float* w_dkn    = (const float*)d_in[7];
  const float* w_dv     = (const float*)d_in[8];
  const float* w_krope  = (const float*)d_in[9];
  const float* w_imp    = (const float*)d_in[10];
  const float* w_selk   = (const float*)d_in[11];
  const float* w_selv   = (const float*)d_in[12];
  const float* w_wink   = (const float*)d_in[13];
  const float* w_winv   = (const float*)d_in[14];
  const float* w_gate   = (const float*)d_in[15];
  const float* w_proj   = (const float*)d_in[16];
  float* out = (float*)d_out;

  const int BT = CB * CT;          // 2048
  const int BS = CB * CKEEP;       // 512

  size_t off = 0;
  auto wsa = [&](size_t bytes) -> void* {
    off = (off + 255) & ~(size_t)255;
    void* p = (void*)((char*)d_ws + off);
    off += bytes;
    return p;
  };

  // f16 buffers
  _Float16* x_h    = (_Float16*)wsa((size_t)BT * CC * 2);
  _Float16* nq_h   = (_Float16*)wsa((size_t)BT * CQLORA * 2);
  _Float16* ckv_h  = (_Float16*)wsa((size_t)BT * CKVLORA * 2);
  _Float16* sel_h  = (_Float16*)wsa((size_t)BS * CC * 2);
  _Float16* Q_h    = (_Float16*)wsa((size_t)CB * CH * CT * CQD * 2);
  _Float16* K1_h   = (_Float16*)wsa((size_t)CB * CH * CT * CQD * 2);
  _Float16* V1_h   = (_Float16*)wsa((size_t)CB * CH * CT * CVH * 2);
  _Float16* Ks_h   = (_Float16*)wsa((size_t)CB * CH * CKEEP * CQD * 2);
  _Float16* Vs_h   = (_Float16*)wsa((size_t)CB * CH * CKEEP * CVH * 2);
  _Float16* Kw_h   = (_Float16*)wsa((size_t)CB * CH * CT * CQD * 2);
  _Float16* Vw_h   = (_Float16*)wsa((size_t)CB * CH * CT * CVH * 2);
  _Float16* attn_h = (_Float16*)wsa((size_t)BT * (CH * CVH) * 2);
  // f32 buffers
  float* qn_raw = (float*)wsa((size_t)BT * (CH * CNOPE) * 4);
  float* qr_raw = (float*)wsa((size_t)BT * (CH * CRD) * 4);
  float* kn_raw = (float*)wsa((size_t)BT * (CH * CNOPE) * 4);
  float* v1_raw = (float*)wsa((size_t)BT * (CH * CVH) * 4);
  float* kr_raw = (float*)wsa((size_t)BT * CRD * 4);
  float* ks_raw = (float*)wsa((size_t)BS * (CH * CQD) * 4);
  float* vs_raw = (float*)wsa((size_t)BS * (CH * CVH) * 4);
  float* kw_raw = (float*)wsa((size_t)BT * (CH * CQD) * 4);
  float* vw_raw = (float*)wsa((size_t)BT * (CH * CVH) * 4);
  float* scores = (float*)wsa((size_t)BT * 4);
  float* glog   = (float*)wsa((size_t)BT * 3 * 4);
  float* gate   = (float*)wsa((size_t)CB * 3 * 4);
  float* O1     = (float*)wsa((size_t)CB * CH * CT * CVH * 4);
  float* O2     = (float*)wsa((size_t)CB * CH * CT * CVH * 4);
  float* O3     = (float*)wsa((size_t)CB * CH * CT * CVH * 4);
  int*   idx    = (int*)wsa((size_t)CB * CKEEP * 4);

  // 1) small projections + norms
  smallproj_kernel<<<BT, 160, 0, stream>>>(x, w_cq, g_qnorm, w_ckv, g_kvnorm,
                                           w_imp, w_gate, x_h, nq_h, ckv_h,
                                           scores, glog);
  // 2) gate + top-k + gather
  gate_kernel<<<CB, 256, 0, stream>>>(glog, gate, CT);
  topk_kernel<<<CB, CT, 0, stream>>>(scores, idx, CT);
  gather_kernel<<<BS, 256, 0, stream>>>(x, idx, sel_h);

  // 3) projection GEMMs (WMMA, 64x64 tiles, 4 waves/block)
  auto gemm = [&](const _Float16* A, int lda, const float* Bw, int ldb,
                  float* Cmat, int ldc, int M, int N, int K, float sc) {
    gemm_wmma_kernel<<<dim3(N / 64, M / 64), 128, 0, stream>>>(A, lda, Bw, ldb,
                                                               Cmat, ldc, K, sc);
  };
  gemm(nq_h, CQLORA, w_dqn, CH * CNOPE, qn_raw, CH * CNOPE, BT, CH * CNOPE, CQLORA, 1.f);
  gemm(nq_h, CQLORA, w_dqr, CH * CRD,   qr_raw, CH * CRD,   BT, CH * CRD,   CQLORA, 1.f);
  gemm(ckv_h, CKVLORA, w_dkn, CH * CNOPE, kn_raw, CH * CNOPE, BT, CH * CNOPE, CKVLORA, 1.f);
  gemm(ckv_h, CKVLORA, w_dv,  CH * CVH,   v1_raw, CH * CVH,   BT, CH * CVH,   CKVLORA, 1.f);
  gemm(x_h, CC, w_krope, CRD, kr_raw, CRD, BT, CRD, CC, 1.0f / (float)CH);
  gemm(sel_h, CC, w_selk, CH * CQD, ks_raw, CH * CQD, BS, CH * CQD, CC, 1.f);
  gemm(sel_h, CC, w_selv, CH * CVH, vs_raw, CH * CVH, BS, CH * CVH, CC, 1.f);
  gemm(x_h, CC, w_wink, CH * CQD, kw_raw, CH * CQD, BT, CH * CQD, CC, 1.f);
  gemm(x_h, CC, w_winv, CH * CVH, vw_raw, CH * CVH, BT, CH * CVH, CC, 1.f);

  // 4) rope + pack into [B,H,S,D] f16
  pack_q_kernel<<<BT * CH, 64, 0, stream>>>(qn_raw, qr_raw, Q_h);
  pack_k1_kernel<<<BT * CH, 96, 0, stream>>>(kn_raw, kr_raw, v1_raw, K1_h, V1_h);
  pack_skw_kernel<<<BS * CH, 96, 0, stream>>>(ks_raw, vs_raw, Ks_h, Vs_h, CKEEP);
  pack_skw_kernel<<<BT * CH, 96, 0, stream>>>(kw_raw, vw_raw, Kw_h, Vw_h, CT);

  // 5) three attention branches (flash, WMMA)
  dim3 agrid(CT / 16, CH, CB);
  attn_kernel<<<agrid, 32, 0, stream>>>(Q_h, K1_h, V1_h, O1, CT, CT, 1);
  attn_kernel<<<agrid, 32, 0, stream>>>(Q_h, Ks_h, Vs_h, O2, CT, CKEEP, 0);
  attn_kernel<<<agrid, 32, 0, stream>>>(Q_h, Kw_h, Vw_h, O3, CT, CT, 1);

  // 6) gated combine + final projection
  combine_kernel<<<BT, CH * CVH, 0, stream>>>(O1, O2, O3, gate, attn_h);
  gemm(attn_h, CH * CVH, w_proj, CC, out, CC, BT, CC, CH * CVH, 1.f);
}
